// MultiHeadsDotProductAttentionLayer_22041772163394
// MI455X (gfx1250) — compile-verified, matrix operand order checked by reading, not run on hardware
//
#include <hip/hip_runtime.h>
#include <hip/hip_bf16.h>

// ---------------------------------------------------------------------------
// MultiHeadsDotProductAttention for MI455X (gfx1250, wave32, WMMA bf16)
// B=8, S=512, HIDDEN=1024, N_HEADS=16, D=64
// ---------------------------------------------------------------------------

typedef __attribute__((ext_vector_type(8)))  float   v8f;
typedef __attribute__((ext_vector_type(4)))  float   v4f;
typedef __attribute__((ext_vector_type(16))) __bf16  v16bf;
typedef __attribute__((ext_vector_type(8)))  __bf16  v8bf;
typedef __attribute__((ext_vector_type(4)))  __bf16  v4bf;

union Frag {
    v16bf v;
    v8bf  h[2];
};

__device__ inline v8f zero8() {
    v8f z;
#pragma unroll
    for (int i = 0; i < 8; ++i) z[i] = 0.f;
    return z;
}

__device__ inline v8f wmma_bf16(const Frag& a, const Frag& b, v8f c) {
    // D = A(16x32 bf16) * B(32x16 bf16) + C(16x16 f32)
    return __builtin_amdgcn_wmma_f32_16x16x32_bf16(
        /*neg_a=*/false, a.v, /*neg_b=*/false, b.v,
        /*c_mod=*/(short)0, c, /*reuse_a=*/false, /*reuse_b=*/false);
}

// ---------------------------------------------------------------------------
// Generic WMMA GEMM: C[M,N] = alpha * A[M,K] @ W[K,N]
//   A_BF16=false : A is fp32 (x or y), C written as bf16         (QKV proj)
//   RESID=true   : A is bf16 attention output, epilogue computes
//                  z = C + resid, stores fp32 z and atomically
//                  accumulates per-batch sum(z), sum(z^2)         (Wo proj)
// Block: 128 threads (4 waves), tile 64(M) x 64(N), K-step 32.
// ---------------------------------------------------------------------------
template <bool A_BF16, bool RESID>
__global__ __launch_bounds__(128)
void gemm_kernel(const void* __restrict__ Ap, const float* __restrict__ W,
                 void* __restrict__ Cp, const float* __restrict__ resid,
                 float* __restrict__ zout, float* __restrict__ sums,
                 float* __restrict__ sumsq, int M, int N, int K, float alpha) {
    __shared__ __bf16 Asl[64][32];   // A tile (bf16)
    __shared__ __bf16 Wtl[64][32];   // W tile transposed: Wtl[n][kc] = W[k0+kc][n0+n]
    __shared__ float  r1[128], r2[128];

    const int t    = threadIdx.x;
    const int wave = t >> 5;
    const int lane = t & 31;
    const int lg   = lane & 15;
    const int kb0  = (lane >= 16) ? 8 : 0;   // A/B fragment K-offset per lane half
    const int hi   = (lane >= 16) ? 8 : 0;   // C row offset per lane half
    const int m0   = blockIdx.y * 64;
    const int n0   = blockIdx.x * 64;

    v8f acc[4];
#pragma unroll
    for (int i = 0; i < 4; ++i) acc[i] = zero8();

    for (int k0 = 0; k0 < K; k0 += 32) {
        // stage A tile (vectorized read, convert to bf16, apply alpha)
        for (int i = t; i < 64 * 8; i += 128) {           // 512 x (4-elem chunks)
            const int m = i >> 3, kc4 = (i & 7) * 4;
            v4f  av;
            v4bf ab;
            if (A_BF16) {
                const v4bf raw =
                    *(const v4bf*)&((const __bf16*)Ap)[(size_t)(m0 + m) * K + k0 + kc4];
#pragma unroll
                for (int u = 0; u < 4; ++u) av[u] = (float)raw[u];
            } else {
                av = *(const v4f*)&((const float*)Ap)[(size_t)(m0 + m) * K + k0 + kc4];
            }
#pragma unroll
            for (int u = 0; u < 4; ++u) ab[u] = (__bf16)(alpha * av[u]);
            *(v4bf*)&Asl[m][kc4] = ab;
        }
        // stage W tile transposed (vectorized along n, scatter bf16 to LDS)
        for (int i = t; i < 32 * 16; i += 128) {          // 512 x (4-elem chunks)
            const int kc = i >> 4, n4 = (i & 15) * 4;
            const v4f wv = *(const v4f*)&W[(size_t)(k0 + kc) * N + n0 + n4];
#pragma unroll
            for (int u = 0; u < 4; ++u) Wtl[n4 + u][kc] = (__bf16)wv[u];
        }
        __syncthreads();

        Frag a;
        a.h[0] = *(const v8bf*)&Asl[wave * 16 + lg][kb0];
        a.h[1] = *(const v8bf*)&Asl[wave * 16 + lg][kb0 + 16];
#pragma unroll
        for (int nt = 0; nt < 4; ++nt) {
            Frag b;
            b.h[0] = *(const v8bf*)&Wtl[nt * 16 + lg][kb0];
            b.h[1] = *(const v8bf*)&Wtl[nt * 16 + lg][kb0 + 16];
            acc[nt] = wmma_bf16(a, b, acc[nt]);
        }
        __syncthreads();
    }

    // epilogue
    float lsum = 0.f, lsq = 0.f;
#pragma unroll
    for (int nt = 0; nt < 4; ++nt) {
#pragma unroll
        for (int j = 0; j < 8; ++j) {
            const int   m  = m0 + wave * 16 + hi + j;
            const int   n  = n0 + nt * 16 + lg;
            const float cv = acc[nt][j];
            if constexpr (!RESID) {
                ((__bf16*)Cp)[(size_t)m * N + n] = (__bf16)cv;
            } else {
                const float zv = cv + resid[(size_t)m * N + n];
                zout[(size_t)m * N + n] = zv;
                lsum += zv;
                lsq  += zv * zv;
            }
        }
    }
    if constexpr (RESID) {
        r1[t] = lsum;
        r2[t] = lsq;
        __syncthreads();
        for (int s = 64; s > 0; s >>= 1) {
            if (t < s) { r1[t] += r1[t + s]; r2[t] += r2[t + s]; }
            __syncthreads();
        }
        if (t == 0) {
            atomicAdd(&sums[m0 >> 9],  r1[0]);   // batch = m0 / 512
            atomicAdd(&sumsq[m0 >> 9], r2[0]);
        }
    }
}

// ---------------------------------------------------------------------------
// Fused attention: one block per (q-tile of 64, head*batch).
// scores = exp(Q Kt) * y_mask kept as bf16 in LDS (64x512 = 64KB),
// out = (x_mask/(rowsum+eps)) * (E @ V), attention matrix written to d_out
// in a single normalized non-temporal pass at the end.
// ---------------------------------------------------------------------------
__global__ __launch_bounds__(128)
void attn_kernel(const __bf16* __restrict__ Qb, const __bf16* __restrict__ Kb,
                 const __bf16* __restrict__ Vb, const float* __restrict__ xmask,
                 const float* __restrict__ ymask, float* __restrict__ att_out,
                 __bf16* __restrict__ oattn) {
    __shared__ __bf16 Ebig[64][512];   // exp'd scores, bf16   (64 KB)
    __shared__ __bf16 Vt[64][72];      // V tile transposed: Vt[d][k]
    __shared__ float  denomL[64];

    const int qt = blockIdx.x;          // 0..7
    const int hb = blockIdx.y;          // 0..127  (h*8 + b)
    const int h  = hb >> 3;
    const int b  = hb & 7;
    const int q0 = qt * 64;

    const int t    = threadIdx.x;
    const int wave = t >> 5;
    const int lane = t & 31;
    const int lg   = lane & 15;
    const int kb0  = (lane >= 16) ? 8 : 0;
    const int hi   = (lane >= 16) ? 8 : 0;

    // Q fragments for this wave's 16-row strip (Q pre-scaled by 1/sqrt(64))
    Frag qf[2];
    {
        const __bf16* qp =
            Qb + (size_t)(b * 512 + q0 + wave * 16 + lg) * 1024 + h * 64;
#pragma unroll
        for (int s = 0; s < 2; ++s) {
            qf[s].h[0] = *(const v8bf*)(qp + s * 32 + kb0);
            qf[s].h[1] = *(const v8bf*)(qp + s * 32 + kb0 + 16);
        }
    }

    v8f oacc[4];
#pragma unroll
    for (int i = 0; i < 4; ++i) oacc[i] = zero8();
    float rs[8] = {};

    for (int kt = 0; kt < 8; ++kt) {
        // stage V tile transposed (vectorized global reads, bf16 scatter)
        for (int i = t; i < 64 * 16; i += 128) {          // 1024 x (4-elem chunks)
            const int kk = i >> 4, d4 = (i & 15) * 4;
            const v4bf vv =
                *(const v4bf*)&Vb[(size_t)(b * 512 + kt * 64 + kk) * 1024 + h * 64 + d4];
#pragma unroll
            for (int u = 0; u < 4; ++u) Vt[d4 + u][kk] = vv[u];
        }
        __syncthreads();

        // scores for 64 k-columns: 4 N-tiles x 2 K-steps of WMMA
#pragma unroll
        for (int nt = 0; nt < 4; ++nt) {
            v8f c = zero8();
#pragma unroll
            for (int s = 0; s < 2; ++s) {
                Frag kf;
                const __bf16* kp =
                    Kb + (size_t)(b * 512 + kt * 64 + nt * 16 + lg) * 1024 +
                    h * 64 + s * 32;
                kf.h[0] = *(const v8bf*)(kp + kb0);
                kf.h[1] = *(const v8bf*)(kp + kb0 + 16);
                c = wmma_bf16(qf[s], kf, c);
            }
            const int   col = kt * 64 + nt * 16 + lg;
            const float ym  = ymask[b * 512 + col];
#pragma unroll
            for (int j = 0; j < 8; ++j) {
                const float  e  = __expf(c[j]) * ym;
                const __bf16 eb = (__bf16)e;
                rs[j] += (float)eb;               // rowsum of stored values
                Ebig[wave * 16 + hi + j][col] = eb;
            }
        }

        // partial out += E_strip(16x64) @ V(64x64); LDS ops in-order per wave
#pragma unroll
        for (int s = 0; s < 2; ++s) {
            Frag af;
            const __bf16* ep = &Ebig[wave * 16 + lg][kt * 64 + s * 32];
            af.h[0] = *(const v8bf*)(ep + kb0);
            af.h[1] = *(const v8bf*)(ep + kb0 + 16);
#pragma unroll
            for (int nt = 0; nt < 4; ++nt) {
                Frag vf;
                const __bf16* vp = &Vt[nt * 16 + lg][s * 32];
                vf.h[0] = *(const v8bf*)(vp + kb0);
                vf.h[1] = *(const v8bf*)(vp + kb0 + 16);
                oacc[nt] = wmma_bf16(af, vf, oacc[nt]);
            }
        }
        __syncthreads();   // before next iteration overwrites Vt
    }

    // full row sums via intra-group butterfly (lanes 0-15 / 16-31 hold
    // disjoint column subsets of the same rows)
#pragma unroll
    for (int j = 0; j < 8; ++j) {
        float v = rs[j];
        v += __shfl_xor(v, 1, 16);
        v += __shfl_xor(v, 2, 16);
        v += __shfl_xor(v, 4, 16);
        v += __shfl_xor(v, 8, 16);
        rs[j] = v;
    }
    if (lg == 0) {
#pragma unroll
        for (int j = 0; j < 8; ++j) denomL[wave * 16 + hi + j] = rs[j];
    }
    __syncthreads();

    // scaled attention-output tile -> bf16 workspace (layout [b,q,h*64+d])
#pragma unroll
    for (int j = 0; j < 8; ++j) {
        const int   q   = q0 + wave * 16 + hi + j;
        const float scl = xmask[b * 512 + q] / (rs[j] + 1e-8f);
#pragma unroll
        for (int nt = 0; nt < 4; ++nt)
            oattn[(size_t)(b * 512 + q) * 1024 + h * 64 + nt * 16 + lg] =
                (__bf16)(oacc[nt][j] * scl);
    }

    // single normalized, vectorized, non-temporal write of the attention
    // matrix: 8 columns per chunk (ds_load_b128 + 2x b128 NT global stores)
    for (int i = t; i < 64 * 64; i += 128) {              // 4096 x (8-col chunks)
        const int   r    = i >> 6, c0 = (i & 63) * 8;
        const int   q    = q0 + r;
        const float scl  = xmask[b * 512 + q] / (denomL[r] + 1e-8f);
        const v8bf  ev   = *(const v8bf*)&Ebig[r][c0];
        v4f lo, hiv;
#pragma unroll
        for (int u = 0; u < 4; ++u) {
            lo[u]  = (float)ev[u]     * scl;
            hiv[u] = (float)ev[u + 4] * scl;
        }
        float* dst = &att_out[((size_t)hb * 512 + q) * 512 + c0];
        __builtin_nontemporal_store(lo,  (v4f*)dst);
        __builtin_nontemporal_store(hiv, (v4f*)(dst + 4));
    }
}

// ---------------------------------------------------------------------------
__global__ void init_kernel(float* sums, float* sumsq) {
    if (threadIdx.x < 8) {
        sums[threadIdx.x]  = 0.f;
        sumsq[threadIdx.x] = 0.f;
    }
}

// Layer norm over axes (1,2) per batch; gamma/beta on last axis.
__global__ __launch_bounds__(256)
void ln_kernel(const float* __restrict__ z, const float* __restrict__ sums,
               const float* __restrict__ sumsq, const float* __restrict__ gamma,
               const float* __restrict__ beta, float* __restrict__ out) {
    const size_t i = (size_t)blockIdx.x * 256 + threadIdx.x;
    const int    b = (int)(i >> 19);        // 512*1024 elems per batch
    const int    c = (int)(i & 1023);
    const float  inv  = 1.f / 524288.f;
    const float  mean = sums[b] * inv;
    const float  var  = sumsq[b] * inv - mean * mean;
    out[i] = gamma[c] * (z[i] - mean) * rsqrtf(var + 1e-12f) + beta[c];
}

// ---------------------------------------------------------------------------
extern "C" void kernel_launch(void* const* d_in, const int* in_sizes, int n_in,
                              void* d_out, int out_size, void* d_ws,
                              size_t ws_size, hipStream_t stream) {
    (void)in_sizes; (void)n_in; (void)out_size; (void)ws_size;
    const float* x     = (const float*)d_in[0];
    const float* y     = (const float*)d_in[1];
    const float* xmask = (const float*)d_in[2];
    const float* ymask = (const float*)d_in[3];
    const float* Wq    = (const float*)d_in[4];
    const float* Wk    = (const float*)d_in[5];
    const float* Wv    = (const float*)d_in[6];
    const float* Wo    = (const float*)d_in[7];
    const float* gamma = (const float*)d_in[8];
    const float* beta  = (const float*)d_in[9];
    float* out = (float*)d_out;

    // workspace layout (~48 MB)
    char*   w   = (char*)d_ws;
    __bf16* Qb  = (__bf16*)w;  w += (size_t)4096 * 1024 * 2;
    __bf16* Kb  = (__bf16*)w;  w += (size_t)4096 * 1024 * 2;
    __bf16* Vb  = (__bf16*)w;  w += (size_t)4096 * 1024 * 2;
    __bf16* OA  = (__bf16*)w;  w += (size_t)4096 * 1024 * 2;
    float*  z   = (float*)w;   w += (size_t)4096 * 1024 * 4;
    float*  sums  = (float*)w; w += 256;
    float*  sumsq = (float*)w; w += 256;

    init_kernel<<<1, 32, 0, stream>>>(sums, sumsq);

    const dim3 gblk(128);
    const dim3 ggrid(16, 64);  // N/64, M/64
    // Q pre-scaled by 1/sqrt(D_K) = 0.125
    gemm_kernel<false, false><<<ggrid, gblk, 0, stream>>>(
        x, Wq, Qb, nullptr, nullptr, nullptr, nullptr, 4096, 1024, 1024, 0.125f);
    gemm_kernel<false, false><<<ggrid, gblk, 0, stream>>>(
        y, Wk, Kb, nullptr, nullptr, nullptr, nullptr, 4096, 1024, 1024, 1.0f);
    gemm_kernel<false, false><<<ggrid, gblk, 0, stream>>>(
        y, Wv, Vb, nullptr, nullptr, nullptr, nullptr, 4096, 1024, 1024, 1.0f);

    attn_kernel<<<dim3(8, 128), dim3(128), 0, stream>>>(
        Qb, Kb, Vb, xmask, ymask, out, OA);

    // z = x + OA @ Wo, plus per-batch sum/sumsq for layer norm
    gemm_kernel<true, true><<<ggrid, gblk, 0, stream>>>(
        OA, Wo, nullptr, x, z, sums, sumsq, 4096, 1024, 1024, 1.0f);

    ln_kernel<<<16384, 256, 0, stream>>>(
        z, sums, sumsq, gamma, beta, out + (size_t)128 * 512 * 512);
}
